// Decoder_28887950033170
// MI455X (gfx1250) — matrix operand order repaired and checked by reading, not compiled, not run on hardware
//
#include <hip/hip_runtime.h>
#include <math.h>

// ---------------------------------------------------------------------------
// Problem constants (match reference)
// ---------------------------------------------------------------------------
enum {
  CL   = 6,      // layers
  CD   = 512,    // d_model
  CH   = 8,      // heads
  CDEP = 64,     // head dim
  CF   = 2048,   // ffn dim
  CB   = 8,      // batch
  CS   = 512,    // decoder seq
  CSE  = 512,    // encoder seq
};

typedef __attribute__((ext_vector_type(16))) __bf16 v16bf;
typedef __attribute__((ext_vector_type(2)))  __bf16 bf16x2;
typedef __attribute__((ext_vector_type(8)))  float  v8f;

union FragBF { v16bf v; unsigned int u32[8]; };
union FragC  { v8f   v; float        f[8];   };

// Hardware bf16 convert (RNE): lowers to v_cvt_pk_bf16_f32
__device__ __forceinline__ unsigned int pack2bf(float lo, float hi) {
  bf16x2 p;
  p.x = (__bf16)lo;
  p.y = (__bf16)hi;
  return __builtin_bit_cast(unsigned int, p);
}

// ---------------------------------------------------------------------------
// Generic batched GEMM:  C = alpha * (A x B[^T]) + bias + resid, optional relu
//   block = 128 threads = 4 waves; block tile 64x64; wave tile 32x32
//   TILE_K = 64 -> 8 x v_wmma_f32_16x16x32_bf16 per wave per staging round
//   Staging is split into a batched load phase (all tile loads in flight)
//   and a convert/store phase, so global latency is covered once per tile.
//   LDS layout: A[m][k] and B[n][k], bf16-packed dwords, row stride 36 dwords
//   (144 B: 16B-aligned for ds_load_b128, 36r mod 64 -> conflict-free banks)
//   batch index z decomposed as (zb, zh) = (z / Hdiv, z % Hdiv)
// Requires: M%64==0, N%64==0, K%64==0, lda/ldb/ldc %4==0 (true for all uses)
// ---------------------------------------------------------------------------
#define ROWD 36   // dwords per LDS tile row (64 bf16 = 32 dwords + 4 pad)

template<bool TRANS_B>
__global__ __launch_bounds__(128) void gemm_wmma_bf16(
    const float* __restrict__ A, const float* __restrict__ Bm,
    const float* __restrict__ bias, const float* __restrict__ resid,
    float* __restrict__ C,
    int M, int N, int K, int lda, int ldb, int ldc,
    int Hdiv, long sAb, long sAh, long sBb, long sBh, long sCb, long sCh,
    float alpha, int act)
{
  __shared__ unsigned int lA[64 * ROWD];   // A tile: [m][k] bf16-packed
  __shared__ unsigned int lB[64 * ROWD];   // B tile: [n][k] bf16-packed

  const int z  = blockIdx.z;
  const int zb = z / Hdiv;
  const int zh = z - zb * Hdiv;
  A  += (long)zb * sAb + (long)zh * sAh;
  Bm += (long)zb * sBb + (long)zh * sBh;
  C  += (long)zb * sCb + (long)zh * sCh;
  if (resid) resid += (long)zb * sCb + (long)zh * sCh;

  const int tileM = blockIdx.y * 64;
  const int tileN = blockIdx.x * 64;
  const int tid   = threadIdx.x;
  const int lane  = tid & 31;
  const int wave  = tid >> 5;
  const int wm0   = (wave >> 1) * 32;   // wave row origin in block tile
  const int wn0   = (wave & 1) * 32;    // wave col origin
  const int lh    = lane & 15;
  const int hi    = lane >> 4;          // 0: lanes 0-15, 1: lanes 16-31

  // Per-thread staging coordinates (invariant across K loop)
  const int aRow = tid >> 4;            // 16 float4 per A row
  const int aC4  = (tid & 15) << 2;
  const int bN   = tid & 63;            // NN path: n per thread
  const int bK4g = (tid >> 6) << 2;     // NN path: k4 group base (0 or 4)

  FragC acc[2][2];
#pragma unroll
  for (int i = 0; i < 2; ++i)
#pragma unroll
    for (int j = 0; j < 2; ++j)
#pragma unroll
      for (int r = 0; r < 8; ++r) acc[i][j].f[r] = 0.0f;

  for (int k0 = 0; k0 < K; k0 += 64) {
    float4 ra[8], rb[8];
    // ---- phase 1: issue ALL tile loads back-to-back ----
#pragma unroll
    for (int it = 0; it < 8; ++it) {
      int v   = tid + it * 128;
      int row = v >> 4;
      int c4  = (v & 15) << 2;
      ra[it] = *(const float4*)(A + (long)(tileM + row) * lda + k0 + c4);
    }
    if (TRANS_B) {
      // global B is [N][K]: contiguous along K, same pattern as A
#pragma unroll
      for (int it = 0; it < 8; ++it) {
        int v  = tid + it * 128;
        int n  = v >> 4;
        int c4 = (v & 15) << 2;
        rb[it] = *(const float4*)(Bm + (long)(tileN + n) * ldb + k0 + c4);
      }
    } else {
      // global B is [K][N]: 4 floats down a K column per item (coalesced
      // across lanes: adjacent lanes cover adjacent n)
#pragma unroll
      for (int it = 0; it < 8; ++it) {
        int k4 = bK4g + (it & 1) * 4 + (it >> 1) * 8;   // covers 0..32
        const float* gp = Bm + (long)(k0 + k4) * ldb + tileN + bN;
        rb[it].x = gp[0];
        rb[it].y = gp[ldb];
        rb[it].z = gp[2 * (long)ldb];
        rb[it].w = gp[3 * (long)ldb];
      }
    }
    // prefetch next A K-chunk (global_prefetch_b8)
    if (k0 + 64 < K)
      __builtin_prefetch(A + (long)(tileM + (tid >> 1)) * lda + k0 + 64, 0, 1);

    // ---- phase 2: convert + store to LDS ----
#pragma unroll
    for (int it = 0; it < 8; ++it) {
      int row = aRow + it * 8;          // (tid + it*128) >> 4
      uint2 p; p.x = pack2bf(ra[it].x, ra[it].y); p.y = pack2bf(ra[it].z, ra[it].w);
      *(uint2*)&lA[row * ROWD + (aC4 >> 1)] = p;
    }
    if (TRANS_B) {
#pragma unroll
      for (int it = 0; it < 8; ++it) {
        int n = aRow + it * 8;
        uint2 p; p.x = pack2bf(rb[it].x, rb[it].y); p.y = pack2bf(rb[it].z, rb[it].w);
        *(uint2*)&lB[n * ROWD + (aC4 >> 1)] = p;
      }
    } else {
#pragma unroll
      for (int it = 0; it < 8; ++it) {
        int k4 = bK4g + (it & 1) * 4 + (it >> 1) * 8;
        uint2 p; p.x = pack2bf(rb[it].x, rb[it].y); p.y = pack2bf(rb[it].z, rb[it].w);
        *(uint2*)&lB[bN * ROWD + (k4 >> 1)] = p;
      }
    }
    __syncthreads();

    // ---- 2 k-sub-steps of 32, fragments are contiguous ds_load_b128 ----
#pragma unroll
    for (int ks = 0; ks < 2; ++ks) {
      const int kd = ks * 16;           // dword offset of this k-sub-chunk
      FragBF a[2], b[2];
#pragma unroll
      for (int mf = 0; mf < 2; ++mf) {
        int row = wm0 + mf * 16 + lh;
        const unsigned int* pa = &lA[row * ROWD + kd];
        *(uint4*)&a[mf].u32[0] = *(const uint4*)&pa[hi * 8];       // K 0-7 / 16-23
        *(uint4*)&a[mf].u32[4] = *(const uint4*)&pa[hi * 8 + 4];   // K 8-15 / 24-31
      }
#pragma unroll
      for (int nf = 0; nf < 2; ++nf) {
        int col = wn0 + nf * 16 + lh;
        const unsigned int* pb = &lB[col * ROWD + kd + hi * 8];    // K 0-15 / 16-31
        *(uint4*)&b[nf].u32[0] = *(const uint4*)&pb[0];
        *(uint4*)&b[nf].u32[4] = *(const uint4*)&pb[4];
      }
#pragma unroll
      for (int mf = 0; mf < 2; ++mf)
#pragma unroll
        for (int nf = 0; nf < 2; ++nf)
          acc[mf][nf].v = __builtin_amdgcn_wmma_f32_16x16x32_bf16(
              false, a[mf].v, false, b[nf].v, (short)0, acc[mf][nf].v, false, false);
    }
    __syncthreads();
  }

  // ---- epilogue: alpha, bias, residual, relu ----
#pragma unroll
  for (int mf = 0; mf < 2; ++mf) {
#pragma unroll
    for (int nf = 0; nf < 2; ++nf) {
      int col  = tileN + wn0 + nf * 16 + lh;
      float bv = bias ? bias[col] : 0.0f;
#pragma unroll
      for (int r = 0; r < 8; ++r) {
        int row  = tileM + wm0 + mf * 16 + hi * 8 + r;      // VGPR r -> M=r / r+8
        float vo = alpha * acc[mf][nf].f[r] + bv;
        if (resid) vo += resid[(long)row * ldc + col];
        if (act == 1) vo = fmaxf(vo, 0.0f);
        C[(long)row * ldc + col] = vo;
      }
    }
  }
}

// ---------------------------------------------------------------------------
// Embedding + sinusoidal positional encoding
// ---------------------------------------------------------------------------
__global__ __launch_bounds__(256) void embed_pe_kernel(
    const int* __restrict__ tokens, const float* __restrict__ emb,
    float* __restrict__ X)
{
  long idx = (long)blockIdx.x * blockDim.x + threadIdx.x;
  const long total = (long)CB * CS * CD;
  if (idx >= total) return;
  int  d   = (int)(idx % CD);
  long bs  = idx / CD;
  int  s   = (int)(bs % CS);
  int  tok = tokens[bs];
  float expo  = (float)(2 * (d / 2)) / (float)CD;
  float denom = powf(10000.0f, expo);
  float angle = (float)s / denom;
  float pe    = ((d & 1) == 0) ? sinf(angle) : cosf(angle);
  X[idx] = emb[(long)tok * CD + d] * sqrtf((float)CD) + pe;
}

// ---------------------------------------------------------------------------
// Masked softmax over last dim. mode 0: causal (col > q masked).
// mode 1: padding mask pmask[b*ncols + col] * (-1e9).
// ---------------------------------------------------------------------------
__global__ __launch_bounds__(256) void softmax_kernel(
    float* __restrict__ logits, const float* __restrict__ pmask,
    int ncols, int Sq, int Hh, int mode)
{
  const int row = blockIdx.x;                 // 0 .. B*H*Sq-1
  const int q   = row % Sq;
  const int b   = row / (Hh * Sq);
  float* ptr = logits + (long)row * ncols;
  const int t = threadIdx.x;
  __shared__ float red[256];

  float m = -1e30f;
  for (int c = t; c < ncols; c += 256) {
    float v = ptr[c];
    if (mode == 0) { if (c > q) v += -1e9f; }
    else           { v += pmask[(long)b * ncols + c] * (-1e9f); }
    ptr[c] = v;
    m = fmaxf(m, v);
  }
  red[t] = m; __syncthreads();
  for (int k = 128; k > 0; k >>= 1) { if (t < k) red[t] = fmaxf(red[t], red[t + k]); __syncthreads(); }
  m = red[0]; __syncthreads();

  float sum = 0.0f;
  for (int c = t; c < ncols; c += 256) {
    float e = __expf(ptr[c] - m);
    ptr[c] = e;
    sum += e;
  }
  red[t] = sum; __syncthreads();
  for (int k = 128; k > 0; k >>= 1) { if (t < k) red[t] += red[t + k]; __syncthreads(); }
  float inv = 1.0f / red[0];
  for (int c = t; c < ncols; c += 256) ptr[c] *= inv;
}

// ---------------------------------------------------------------------------
// LayerNorm over last dim (D=512). One block per row.
// ---------------------------------------------------------------------------
__global__ __launch_bounds__(256) void layernorm_kernel(
    const float* __restrict__ in, const float* __restrict__ g,
    const float* __restrict__ be, float* __restrict__ out, int Dd)
{
  const int row = blockIdx.x;
  const float* p = in + (long)row * Dd;
  float* o = out + (long)row * Dd;
  const int t = threadIdx.x;
  __shared__ float r1[256], r2[256];

  float s = 0.0f, ss = 0.0f;
  for (int c = t; c < Dd; c += 256) { float v = p[c]; s += v; ss += v * v; }
  r1[t] = s; r2[t] = ss; __syncthreads();
  for (int k = 128; k > 0; k >>= 1) {
    if (t < k) { r1[t] += r1[t + k]; r2[t] += r2[t + k]; }
    __syncthreads();
  }
  float mu  = r1[0] / (float)Dd;
  float var = r2[0] / (float)Dd - mu * mu;
  float rs  = rsqrtf(var + 1e-6f);
  for (int c = t; c < Dd; c += 256) o[c] = (p[c] - mu) * rs * g[c] + be[c];
}

// ---------------------------------------------------------------------------
// Host-side GEMM dispatch helper
// ---------------------------------------------------------------------------
static void run_gemm(hipStream_t stream, bool transB,
                     const float* A, const float* Bm, const float* bias,
                     const float* resid, float* C,
                     int M, int N, int K, int lda, int ldb, int ldc,
                     int batches, int Hdiv,
                     long sAb, long sAh, long sBb, long sBh, long sCb, long sCh,
                     float alpha, int act)
{
  dim3 grid((unsigned)(N / 64), (unsigned)(M / 64), (unsigned)batches);
  dim3 blk(128);
  if (transB)
    gemm_wmma_bf16<true ><<<grid, blk, 0, stream>>>(A, Bm, bias, resid, C, M, N, K,
        lda, ldb, ldc, Hdiv, sAb, sAh, sBb, sBh, sCb, sCh, alpha, act);
  else
    gemm_wmma_bf16<false><<<grid, blk, 0, stream>>>(A, Bm, bias, resid, C, M, N, K,
        lda, ldb, ldc, Hdiv, sAb, sAh, sBb, sBh, sCb, sCh, alpha, act);
}

// ---------------------------------------------------------------------------
// kernel_launch
// ---------------------------------------------------------------------------
extern "C" void kernel_launch(void* const* d_in, const int* in_sizes, int n_in,
                              void* d_out, int out_size, void* d_ws, size_t ws_size,
                              hipStream_t stream) {
  (void)in_sizes; (void)n_in; (void)out_size; (void)ws_size;

  const int*   tokens   = (const int*)  d_in[0];
  const float* enc      = (const float*)d_in[1];
  /* d_in[2] look_ahead_mask: causal mask computed analytically */
  const float* pad_mask = (const float*)d_in[3];
  const float* emb      = (const float*)d_in[4];
  const float* s_wq = (const float*)d_in[5],  *s_wk = (const float*)d_in[6];
  const float* s_wv = (const float*)d_in[7],  *s_wo = (const float*)d_in[8];
  const float* s_bq = (const float*)d_in[9],  *s_bk = (const float*)d_in[10];
  const float* s_bv = (const float*)d_in[11], *s_bo = (const float*)d_in[12];
  const float* c_wq = (const float*)d_in[13], *c_wk = (const float*)d_in[14];
  const float* c_wv = (const float*)d_in[15], *c_wo = (const float*)d_in[16];
  const float* c_bq = (const float*)d_in[17], *c_bk = (const float*)d_in[18];
  const float* c_bv = (const float*)d_in[19], *c_bo = (const float*)d_in[20];
  const float* f_w1 = (const float*)d_in[21], *f_b1 = (const float*)d_in[22];
  const float* f_w2 = (const float*)d_in[23], *f_b2 = (const float*)d_in[24];
  const float* l1g  = (const float*)d_in[25], *l1b  = (const float*)d_in[26];
  const float* l2g  = (const float*)d_in[27], *l2b  = (const float*)d_in[28];
  const float* l3g  = (const float*)d_in[29], *l3b  = (const float*)d_in[30];

  float* ws = (float*)d_ws;
  const long NA = (long)CB * CS * CD;          // 2,097,152 floats per act buf
  float* X  = ws + 0 * NA;                     // layer input / residual
  float* Qb = ws + 1 * NA;                     // Q / attn-out scratch
  float* Kb = ws + 2 * NA;
  float* Vb = ws + 3 * NA;
  float* T1 = ws + 4 * NA;                     // pre-layernorm scratch
  float* O1 = ws + 5 * NA;                     // out1
  float* Hb = ws + 6 * NA;                     // FFN hidden [B,S,F]
  float* LG = ws + 6 * NA + (long)CB * CS * CF;// logits [B,H,S,S]

  const int   MR   = CB * CS;                  // 4096 token rows
  const long  sQK  = (long)CS * CD;            // per-batch stride in Q/K/V
  const long  sLGb = (long)CH * CS * CS;       // per-b stride in logits
  const long  sLGh = (long)CS * CS;            // per-h stride in logits
  const float isq  = 0.125f;                   // 1/sqrt(64)

  // ---- embedding + positional encoding ----
  embed_pe_kernel<<<(unsigned)((NA + 255) / 256), 256, 0, stream>>>(tokens, emb, X);

  for (int i = 0; i < CL; ++i) {
    const long oDD = (long)i * CD * CD, oD = (long)i * CD;
    const long oDF = (long)i * CD * CF, oF = (long)i * CF;
    const long oFD = (long)i * CF * CD;

    // ================= self-attention =================
    run_gemm(stream, false, X, s_wq + oDD, s_bq + oD, nullptr, Qb,
             MR, CD, CD, CD, CD, CD, 1, 1, 0,0,0,0,0,0, 1.0f, 0);
    run_gemm(stream, false, X, s_wk + oDD, s_bk + oD, nullptr, Kb,
             MR, CD, CD, CD, CD, CD, 1, 1, 0,0,0,0,0,0, 1.0f, 0);
    run_gemm(stream, false, X, s_wv + oDD, s_bv + oD, nullptr, Vb,
             MR, CD, CD, CD, CD, CD, 1, 1, 0,0,0,0,0,0, 1.0f, 0);
    // logits[b,h] = 0.125 * Q[b,:,h*64:] x K[b,:,h*64:]^T
    run_gemm(stream, true, Qb, Kb, nullptr, nullptr, LG,
             CS, CS, CDEP, CD, CD, CS, CB * CH, CH,
             sQK, CDEP, sQK, CDEP, sLGb, sLGh, isq, 0);
    softmax_kernel<<<(unsigned)(CB * CH * CS), 256, 0, stream>>>(LG, nullptr, CS, CS, CH, 0);
    // attn-out[b,:,h*64:] = probs[b,h] x V[b,:,h*64:]   (reuses Qb)
    run_gemm(stream, false, LG, Vb, nullptr, nullptr, Qb,
             CS, CDEP, CS, CS, CD, CD, CB * CH, CH,
             sLGb, sLGh, sQK, CDEP, sQK, CDEP, 1.0f, 0);
    // out proj + residual(X)
    run_gemm(stream, false, Qb, s_wo + oDD, s_bo + oD, X, T1,
             MR, CD, CD, CD, CD, CD, 1, 1, 0,0,0,0,0,0, 1.0f, 0);
    layernorm_kernel<<<(unsigned)MR, 256, 0, stream>>>(T1, l1g + oD, l1b + oD, O1, CD);

    // ================= cross-attention =================
    run_gemm(stream, false, O1, c_wq + oDD, c_bq + oD, nullptr, Qb,
             MR, CD, CD, CD, CD, CD, 1, 1, 0,0,0,0,0,0, 1.0f, 0);
    run_gemm(stream, false, enc, c_wk + oDD, c_bk + oD, nullptr, Kb,
             CB * CSE, CD, CD, CD, CD, CD, 1, 1, 0,0,0,0,0,0, 1.0f, 0);
    run_gemm(stream, false, enc, c_wv + oDD, c_bv + oD, nullptr, Vb,
             CB * CSE, CD, CD, CD, CD, CD, 1, 1, 0,0,0,0,0,0, 1.0f, 0);
    run_gemm(stream, true, Qb, Kb, nullptr, nullptr, LG,
             CS, CSE, CDEP, CD, CD, CSE, CB * CH, CH,
             sQK, CDEP, (long)CSE * CD, CDEP, sLGb, sLGh, isq, 0);
    softmax_kernel<<<(unsigned)(CB * CH * CS), 256, 0, stream>>>(LG, pad_mask, CSE, CS, CH, 1);
    run_gemm(stream, false, LG, Vb, nullptr, nullptr, Qb,
             CS, CDEP, CSE, CSE, CD, CD, CB * CH, CH,
             sLGb, sLGh, (long)CSE * CD, CDEP, sQK, CDEP, 1.0f, 0);
    run_gemm(stream, false, Qb, c_wo + oDD, c_bo + oD, O1, T1,
             MR, CD, CD, CD, CD, CD, 1, 1, 0,0,0,0,0,0, 1.0f, 0);
    layernorm_kernel<<<(unsigned)MR, 256, 0, stream>>>(T1, l2g + oD, l2b + oD, X, CD); // X = out2

    // ================= FFN =================
    run_gemm(stream, false, X, f_w1 + oDF, f_b1 + oF, nullptr, Hb,
             MR, CF, CD, CD, CF, CF, 1, 1, 0,0,0,0,0,0, 1.0f, 1 /*relu*/);
    run_gemm(stream, false, Hb, f_w2 + oFD, f_b2 + oD, X /*resid out2*/, T1,
             MR, CD, CF, CF, CD, CD, 1, 1, 0,0,0,0,0,0, 1.0f, 0);
    float* dst = (i == CL - 1) ? (float*)d_out : X;
    layernorm_kernel<<<(unsigned)MR, 256, 0, stream>>>(T1, l3g + oD, l3b + oD, dst, CD);
  }
}
